// Loung_NMT_34419867910275
// MI455X (gfx1250) — matrix operand order repaired and checked by reading, not compiled, not run on hardware
//
#include <hip/hip_runtime.h>
#include <hip/hip_bf16.h>

// ---------------------------------------------------------------------------
// Luong NMT on gfx1250 (MI455X). All GEMMs via v_wmma_f32_16x16x32_f16.
// Key algebraic simplification: classifier "hidden" == per-step "feed"
// (identical formula), so we save feed in f16 and skip the recompute.
// ---------------------------------------------------------------------------

typedef __attribute__((ext_vector_type(16))) _Float16 v16h;
typedef __attribute__((ext_vector_type(8)))  _Float16 v8h;
typedef __attribute__((ext_vector_type(8)))  float    v8f;

// Problem constants (fixed by the reference).
#define NB 32      // batch
#define NS 100     // source len
#define NT 100     // target len
#define NE 512     // embed dim
#define NH 1024    // hidden dim
#define NG 4096    // 4*H gates
#define NV 32000   // vocab

// ---------------------------------------------------------------------------
// WMMA GEMM:  C[M,N](f32) (+)= A[M,K](f16,row-major,lda) x BT[N,K](f16,row-major,ldbt)^T
//   initMode: 0 -> zero, 1 -> bias[N], 2 -> accumulate into existing C
//   act:      0 -> none, 1 -> tanh
//   out16:    optional second f16 store with row stride ld16 (for feed->hidden)
// One wave computes 4 consecutive 16x16 N-tiles sharing one A fragment.
// Requires: M%16==0, N%256==0, K%32==0 (true for every call here).
// ---------------------------------------------------------------------------
__global__ void wmma_gemm_f16(const _Float16* __restrict__ A, int lda,
                              const _Float16* __restrict__ BT, int ldbt,
                              float* __restrict__ C, int ldc,
                              const float* __restrict__ bias,
                              int initMode, int act,
                              _Float16* __restrict__ out16, int ld16,
                              int M, int N, int K)
{
    const int lane   = threadIdx.x & 31;
    const int wave   = threadIdx.x >> 5;
    const int nTile0 = (blockIdx.x * 4 + wave) * 4;   // 4 waves/block, 4 tiles/wave
    const int mBase  = blockIdx.y * 16;
    const int hs     = lane >> 4;                     // lane-group select
    const int lrow   = lane & 15;

    v8f acc[4];
#pragma unroll
    for (int j = 0; j < 4; ++j) {
        const int nb = (nTile0 + j) * 16 + lrow;
        if (initMode == 2) {
#pragma unroll
            for (int r = 0; r < 8; ++r)
                acc[j][r] = C[(long)(mBase + hs * 8 + r) * ldc + nb];
        } else if (initMode == 1) {
            const float bv = bias[nb];
#pragma unroll
            for (int r = 0; r < 8; ++r) acc[j][r] = bv;
        } else {
#pragma unroll
            for (int r = 0; r < 8; ++r) acc[j][r] = 0.0f;
        }
    }

    // A fragment: row = mBase+lrow; halves 0..15 -> K = hs*16 + (0..15)
    //             (contiguous, matching B)
    const _Float16* aPtr = A + (long)(mBase + lrow) * lda + hs * 16;
    // B fragment from BT: col = lrow of tile; 16 consecutive K per lane group.
    const _Float16* bPtr[4];
#pragma unroll
    for (int j = 0; j < 4; ++j)
        bPtr[j] = BT + (long)((nTile0 + j) * 16 + lrow) * ldbt + hs * 16;

    for (int k0 = 0; k0 < K; k0 += 32) {
        v16h a;
        const v8h alo = *(const v8h*)(aPtr + k0);
        const v8h ahi = *(const v8h*)(aPtr + k0 + 8);
#pragma unroll
        for (int i = 0; i < 8; ++i) { a[i] = alo[i]; a[8 + i] = ahi[i]; }
#pragma unroll
        for (int j = 0; j < 4; ++j) {
            const v16h b = *(const v16h*)(bPtr[j] + k0);
            acc[j] = __builtin_amdgcn_wmma_f32_16x16x32_f16(
                false, a, false, b, (short)0, acc[j], false, false);
        }
    }

#pragma unroll
    for (int j = 0; j < 4; ++j) {
        const int nb = (nTile0 + j) * 16 + lrow;
#pragma unroll
        for (int r = 0; r < 8; ++r) {
            const int row = mBase + hs * 8 + r;
            float v = acc[j][r];
            if (act == 1) v = tanhf(v);
            C[(long)row * ldc + nb] = v;
            if (out16) out16[(long)row * ld16 + nb] = (_Float16)v;
        }
    }
}

// f32 [K,N] row-major  ->  f16 transposed [N,K] row-major
__global__ void conv_transpose_f16(const float* __restrict__ in,
                                   _Float16* __restrict__ out, int K, int N)
{
    const long id = (long)blockIdx.x * blockDim.x + threadIdx.x;
    if (id >= (long)N * K) return;
    const int  k = (int)(id % K);
    const long n = id / K;
    out[n * K + k] = (_Float16)in[(long)k * N + n];
}

// Embedding gather -> f16, time-major rows (t*NB + b)
__global__ void gather_embed(const int* __restrict__ ids,      // [NB, L]
                             const float* __restrict__ emb,    // [V, NE]
                             _Float16* __restrict__ out, int L)
{
    const long id = (long)blockIdx.x * blockDim.x + threadIdx.x; // L*NB*NE
    const int  e  = (int)(id & (NE - 1));
    const long rb = id >> 9;           // NE = 512
    const int  b  = (int)(rb & (NB - 1));
    const int  t  = (int)(rb >> 5);    // NB = 32
    const int  tok = ids[b * L + t];
    out[id] = (_Float16)emb[(long)tok * NE + e];
}

__global__ void zero_state(float* __restrict__ c, _Float16* __restrict__ h0)
{
    const int id = blockIdx.x * blockDim.x + threadIdx.x;  // NB*NH
    c[id]  = 0.0f;
    h0[id] = (_Float16)0.0f;
}

// LSTM pointwise: gates [NB,4H] (i,f,g,o) -> c (in place), h32, h16
__global__ void lstm_gates(const float* __restrict__ g, float* __restrict__ c,
                           float* __restrict__ h32, _Float16* __restrict__ h16)
{
    const int id = blockIdx.x * blockDim.x + threadIdx.x;  // NB*NH
    const int b  = id >> 10;
    const int j  = id & (NH - 1);
    const float* gb = g + (long)b * NG;
    const float iv = gb[j], fv = gb[NH + j], gv = gb[2 * NH + j], ov = gb[3 * NH + j];
    const float si = 1.0f / (1.0f + __expf(-iv));
    const float sf = 1.0f / (1.0f + __expf(-fv));
    const float so = 1.0f / (1.0f + __expf(-ov));
    const float cn = sf * c[id] + si * tanhf(gv);
    const float hn = so * tanhf(cn);
    c[id] = cn; h32[id] = hn; h16[id] = (_Float16)hn;
}

// scores[b,s] = dot(enc_proj[s*NB+b, :], h[b, :])   (one wave per (b,s))
__global__ void attn_scores(const float* __restrict__ ep,   // [NS*NB, NH]
                            const float* __restrict__ h,    // [NB, NH]
                            float* __restrict__ sc)         // [NB*NS] b-major
{
    const int w    = (blockIdx.x * blockDim.x + threadIdx.x) >> 5;
    const int lane = threadIdx.x & 31;
    const int b = w / NS, s = w % NS;
    const float* e  = ep + (long)(s * NB + b) * NH;
    const float* hb = h + (long)b * NH;
    float sum = 0.0f;
    for (int k = lane; k < NH; k += 32) sum += e[k] * hb[k];
#pragma unroll
    for (int off = 16; off; off >>= 1) sum += __shfl_xor(sum, off, 32);
    if (lane == 0) sc[b * NS + s] = sum;
}

__global__ void attn_softmax(const float* __restrict__ sc, float* __restrict__ w)
{
    __shared__ float buf[128];
    const int b = blockIdx.x, t = threadIdx.x;
    const float v = (t < NS) ? sc[b * NS + t] : -3.4e38f;
    buf[t] = v; __syncthreads();
    for (int off = 64; off; off >>= 1) {
        if (t < off) buf[t] = fmaxf(buf[t], buf[t + off]);
        __syncthreads();
    }
    const float m = buf[0]; __syncthreads();
    const float e = (t < NS) ? __expf(v - m) : 0.0f;
    buf[t] = e; __syncthreads();
    for (int off = 64; off; off >>= 1) {
        if (t < off) buf[t] += buf[t + off];
        __syncthreads();
    }
    const float inv = 1.0f / buf[0];
    if (t < NS) w[b * NS + t] = e * inv;
}

// atten[b,:] = sum_s w[b,s]*enc_hs[s*NB+b,:]; build concat(atten,h) in f16
__global__ void attn_combine(const float* __restrict__ wgt,  // [NB*NS]
                             const float* __restrict__ hs,   // [NS*NB, NH] f32
                             const float* __restrict__ h,    // [NB, NH]
                             _Float16* __restrict__ cat)     // [NB, 2*NH]
{
    const int id = blockIdx.x * blockDim.x + threadIdx.x;    // NB*NH
    const int b  = id >> 10;
    const int j  = id & (NH - 1);
    const float* wb = wgt + b * NS;
    float a = 0.0f;
    for (int s = 0; s < NS; ++s) a += wb[s] * hs[(long)(s * NB + b) * NH + j];
    cat[(long)b * (2 * NH) + j]      = (_Float16)a;
    cat[(long)b * (2 * NH) + NH + j] = (_Float16)h[id];
}

// ---------------------------------------------------------------------------
extern "C" void kernel_launch(void* const* d_in, const int* in_sizes, int n_in,
                              void* d_out, int out_size, void* d_ws, size_t ws_size,
                              hipStream_t stream)
{
    (void)in_sizes; (void)n_in; (void)out_size; (void)ws_size;

    const int*   src_ids = (const int*)d_in[0];
    const int*   tgt_ids = (const int*)d_in[1];
    const float* src_emb = (const float*)d_in[2];
    const float* tgt_emb = (const float*)d_in[3];
    const float* enc_Wih = (const float*)d_in[4];
    const float* enc_Whh = (const float*)d_in[5];
    const float* enc_b   = (const float*)d_in[6];
    const float* dec_Wih = (const float*)d_in[7];
    const float* dec_Whh = (const float*)d_in[8];
    const float* dec_b   = (const float*)d_in[9];
    const float* att1_W  = (const float*)d_in[10];
    const float* att1_b  = (const float*)d_in[11];
    const float* cls1_W  = (const float*)d_in[12];
    const float* cls1_b  = (const float*)d_in[13];
    const float* cls2_W  = (const float*)d_in[14];
    const float* cls2_b  = (const float*)d_in[15];
    float* out = (float*)d_out;

    // Workspace carve-up (256B aligned).
    size_t off = 0;
    auto alloc = [&](size_t bytes) -> char* {
        char* p = (char*)d_ws + off;
        off += (bytes + 255) & ~(size_t)255;
        return p;
    };
    _Float16* encWihT = (_Float16*)alloc((size_t)NG * NE * 2);  // [4096,512]
    _Float16* encWhhT = (_Float16*)alloc((size_t)NG * NH * 2);  // [4096,1024]
    _Float16* decWihT = (_Float16*)alloc((size_t)NG * (NE + NH) * 2); // [4096,1536]
    _Float16* decWhhT = (_Float16*)alloc((size_t)NG * NH * 2);
    _Float16* att1T   = (_Float16*)alloc((size_t)NH * NH * 2);  // [1024,1024]
    _Float16* cls1T   = (_Float16*)alloc((size_t)NH * 2 * NH * 2); // [1024,2048]
    _Float16* cls2T   = (_Float16*)alloc((size_t)NV * NH * 2);  // [32000,1024]
    _Float16* srcX    = (_Float16*)alloc((size_t)NS * NB * NE * 2);
    _Float16* tgtX    = (_Float16*)alloc((size_t)NT * NB * NE * 2);
    float*    encXg   = (float*)alloc((size_t)NS * NB * NG * 4);
    float*    decXg   = (float*)alloc((size_t)NT * NB * NG * 4);
    _Float16* encHs16 = (_Float16*)alloc((size_t)NS * NB * NH * 2);
    float*    encHs32 = (float*)alloc((size_t)NS * NB * NH * 4);
    float*    encProj = (float*)alloc((size_t)NS * NB * NH * 4);
    _Float16* decHs16 = (_Float16*)alloc((size_t)NT * NB * NH * 2);
    float*    cbuf    = (float*)alloc((size_t)NB * NH * 4);
    _Float16* h0z     = (_Float16*)alloc((size_t)NB * NH * 2);
    float*    dh32    = (float*)alloc((size_t)NB * NH * 4);
    float*    sc      = (float*)alloc((size_t)NB * NS * 4);
    float*    aw      = (float*)alloc((size_t)NB * NS * 4);
    _Float16* cat16   = (_Float16*)alloc((size_t)NB * 2 * NH * 2);
    float*    feed32  = (float*)alloc((size_t)NB * NH * 4);
    _Float16* hidden16= (_Float16*)alloc((size_t)NB * NT * NH * 2); // rows b*NT+t

    auto gemm = [&](const _Float16* A, int lda, const _Float16* BT, int ldbt,
                    float* C, int ldc, const float* bias, int initMode, int act,
                    _Float16* o16, int ld16, int M, int N, int K) {
        dim3 grid(N / 256, M / 16);
        wmma_gemm_f16<<<grid, 128, 0, stream>>>(A, lda, BT, ldbt, C, ldc, bias,
                                                initMode, act, o16, ld16, M, N, K);
    };
    auto cvt = [&](const float* in, _Float16* o, int K, int N) {
        long n = (long)N * K;
        conv_transpose_f16<<<dim3((unsigned)((n + 255) / 256)), 256, 0, stream>>>(in, o, K, N);
    };

    // ---- weights -> f16 transposed ----
    cvt(enc_Wih, encWihT, NE, NG);
    cvt(enc_Whh, encWhhT, NH, NG);
    cvt(dec_Wih, decWihT, NE + NH, NG);
    cvt(dec_Whh, decWhhT, NH, NG);
    cvt(att1_W,  att1T,   NH, NH);
    cvt(cls1_W,  cls1T,   2 * NH, NH);
    cvt(cls2_W,  cls2T,   NH, NV);

    // ---- embedding gathers + state init ----
    gather_embed<<<dim3(NS * NB * NE / 256), 256, 0, stream>>>(src_ids, src_emb, srcX, NS);
    gather_embed<<<dim3(NT * NB * NE / 256), 256, 0, stream>>>(tgt_ids, tgt_emb, tgtX, NT);
    zero_state<<<dim3(NB * NH / 256), 256, 0, stream>>>(cbuf, h0z);

    // ---- encoder: hoist x@Wih + b for all timesteps ----
    gemm(srcX, NE, encWihT, NE, encXg, NG, enc_b, 1, 0, nullptr, 0,
         NS * NB, NG, NE);

    for (int t = 0; t < NS; ++t) {
        const _Float16* hp = t ? (encHs16 + (size_t)(t - 1) * NB * NH) : h0z;
        gemm(hp, NH, encWhhT, NH, encXg + (size_t)t * NB * NG, NG,
             nullptr, 2, 0, nullptr, 0, NB, NG, NH);
        lstm_gates<<<dim3(NB * NH / 256), 256, 0, stream>>>(
            encXg + (size_t)t * NB * NG, cbuf,
            encHs32 + (size_t)t * NB * NH, encHs16 + (size_t)t * NB * NH);
    }

    // ---- enc_proj = enc_output @ att1_W + att1_b ----
    gemm(encHs16, NH, att1T, NH, encProj, NH, att1_b, 1, 0, nullptr, 0,
         NS * NB, NH, NH);

    // ---- decoder: hoist x@Wih[:E] + b for all timesteps ----
    gemm(tgtX, NE, decWihT, NE + NH, decXg, NG, dec_b, 1, 0, nullptr, 0,
         NT * NB, NG, NE);

    for (int t = 0; t < NT; ++t) {
        float* gt = decXg + (size_t)t * NB * NG;
        // += feed_prev @ Wih[E:]   (feed0 = final encoder h)
        const _Float16* fA  = t ? (hidden16 + (size_t)(t - 1) * NH)
                                : (encHs16 + (size_t)(NS - 1) * NB * NH);
        const int       fLd = t ? (NT * NH) : NH;
        gemm(fA, fLd, decWihT + NE, NE + NH, gt, NG, nullptr, 2, 0, nullptr, 0,
             NB, NG, NH);
        // += h_prev @ Whh          (h0 = final encoder h)
        const _Float16* hA = t ? (decHs16 + (size_t)(t - 1) * NB * NH)
                               : (encHs16 + (size_t)(NS - 1) * NB * NH);
        gemm(hA, NH, decWhhT, NH, gt, NG, nullptr, 2, 0, nullptr, 0, NB, NG, NH);
        // gates -> h, c   (c continues from encoder final c)
        lstm_gates<<<dim3(NB * NH / 256), 256, 0, stream>>>(
            gt, cbuf, dh32, decHs16 + (size_t)t * NB * NH);
        // attention
        attn_scores<<<dim3(NB * NS / 4), 128, 0, stream>>>(encProj, dh32, sc);
        attn_softmax<<<dim3(NB), 128, 0, stream>>>(sc, aw);
        attn_combine<<<dim3(NB * NH / 256), 256, 0, stream>>>(aw, encHs32, dh32, cat16);
        // feed = tanh(cat @ cls1_W + b); also stores f16 into hidden16[b*NT+t]
        gemm(cat16, 2 * NH, cls1T, 2 * NH, feed32, NH, cls1_b, 1, 1,
             hidden16 + (size_t)t * NH, NT * NH, NB, NH, 2 * NH);
    }

    // ---- classifier: outs = hidden @ cls2_W + cls2_b  -> d_out [B,T,V] ----
    gemm(hidden16, NH, cls2T, NH, out, NV, cls2_b, 1, 0, nullptr, 0,
         NB * NT, NV, NH);
}
